// MultiHeadSelfAttn_6880537608658
// MI455X (gfx1250) — matrix operand order repaired and checked by reading, not compiled, not run on hardware
//
#include <hip/hip_runtime.h>

typedef __attribute__((ext_vector_type(16))) _Float16 v16h;
typedef __attribute__((ext_vector_type(8)))  _Float16 v8h;
typedef __attribute__((ext_vector_type(4)))  _Float16 v4h;
typedef __attribute__((ext_vector_type(8)))  float    v8f;
typedef __attribute__((ext_vector_type(4)))  float    v4f;

#define BB 2
#define SS 2048
#define EE 1024
#define HH 16
#define DD 64
#define MM (BB*SS)   // 4096 rows for every GEMM

__device__ __forceinline__ v16h cat8(v8h lo, v8h hi) {
  return __builtin_shufflevector(lo, hi, 0,1,2,3,4,5,6,7,8,9,10,11,12,13,14,15);
}

__device__ __forceinline__ v8f wmma_f16(v16h a, v16h b, v8f c) {
  // D = A(16x32 f16) x B(32x16 f16) + C(16x16 f32)
  return __builtin_amdgcn_wmma_f32_16x16x32_f16(false, a, false, b, (short)0, c, false, false);
}

// ---------------------------------------------------------------- fp32 -> f16
__global__ void cvt_f32_f16(const float* __restrict__ in, _Float16* __restrict__ out, int n4) {
  int i = blockIdx.x * blockDim.x + threadIdx.x;
  if (i < n4) {
    v4f x = *(const v4f*)(in + (size_t)i * 4);
    v4h y;
    y[0] = (_Float16)x[0]; y[1] = (_Float16)x[1];
    y[2] = (_Float16)x[2]; y[3] = (_Float16)x[3];
    *(v4h*)(out + (size_t)i * 4) = y;
  }
}

// --------------------------------------------------- Y = X @ W^T + b  (WMMA)
// MODE 0: f16 out, [b,h,s,d]   (Q, K)
// MODE 1: f16 out, [b,h,d,s]   (V, pre-transposed for flash-attn)
// MODE 2: f32 out, [m, n]      (final projection -> d_out)
// AF16:  A operand is f16 (attention output) vs fp32 (raw inputs, cvt in regs)
template<int MODE, bool AF16>
__global__ void gemm_qkvo(const void* __restrict__ Ain,
                          const _Float16* __restrict__ W16,   // f16, row-major [N,K]
                          const float* __restrict__ bias,
                          void* __restrict__ Out) {
  const int lane = threadIdx.x & 31;
  const int wv   = threadIdx.x >> 5;      // 4 waves per block
  const int lan  = lane & 15;
  const int hf   = lane >> 4;             // lane half (A and B: K-interleave)
  const int m_base = blockIdx.y * 64 + wv * 16;
  const int n_base = blockIdx.x * 64;

  const float*    Af = (const float*)Ain;
  const _Float16* Ah = (const _Float16*)Ain;
  const int arow = m_base + lan;

  v8f zero = {0.f,0.f,0.f,0.f,0.f,0.f,0.f,0.f};
  v8f acc[4] = {zero, zero, zero, zero};

  for (int kb = 0; kb < EE; kb += 32) {
    // A operand: lanes 0-15 rows 0-15 K{0..7,16..23}; lanes 16-31 K{8..15,24..31}
    v16h a;
    if (AF16) {
      const _Float16* ap = Ah + (size_t)arow * EE + kb + hf * 8;
      a = cat8(*(const v8h*)ap, *(const v8h*)(ap + 16));
    } else {
      const float* ap = Af + (size_t)arow * EE + kb + hf * 8;
      v4f f0 = *(const v4f*)(ap +  0), f1 = *(const v4f*)(ap +  4);
      v4f f2 = *(const v4f*)(ap + 16), f3 = *(const v4f*)(ap + 20);
      #pragma unroll
      for (int i = 0; i < 4; i++) {
        a[i]      = (_Float16)f0[i];
        a[4 + i]  = (_Float16)f1[i];
        a[8 + i]  = (_Float16)f2[i];
        a[12 + i] = (_Float16)f3[i];
      }
    }
    // B operand: col n per lane; lanes 0-15 K{0..7,16..23}; lanes 16-31 K{8..15,24..31}
    #pragma unroll
    for (int t = 0; t < 4; t++) {
      const _Float16* wp = W16 + (size_t)(n_base + t * 16 + lan) * EE + kb + hf * 8;
      v16h b = cat8(*(const v8h*)wp, *(const v8h*)(wp + 16));
      acc[t] = wmma_f16(a, b, acc[t]);
    }
  }

  #pragma unroll
  for (int t = 0; t < 4; t++) {
    const int n = n_base + t * 16 + lan;
    const float bval = bias[n];
    #pragma unroll
    for (int r = 0; r < 8; r++) {
      const int m = m_base + r + 8 * hf;            // C: row = r + 8*half
      const float val = acc[t][r] + bval;
      if (MODE == 2) {
        ((float*)Out)[(size_t)m * EE + n] = val;
      } else {
        const int bb = m >> 11, s = m & (SS - 1);
        const int h  = n >> 6,  d = n & (DD - 1);
        const _Float16 hv = (_Float16)val;
        if (MODE == 0)
          ((_Float16*)Out)[(((size_t)bb * HH + h) * SS + s) * DD + d] = hv;
        else
          ((_Float16*)Out)[(((size_t)bb * HH + h) * DD + d) * SS + s] = hv;
      }
    }
  }
}

// ----------------------------------------------- causal flash attention core
// Computes O^T = V^T @ P^T with S^T = K @ Q^T; softmax stats live per-lane
// because each lane column of the C tiles is one query. No LDS, no score
// materialization; 8 WMMAs per 32-key block.
__global__ void flash_attn(const _Float16* __restrict__ Q,   // [b,h,s,d]
                           const _Float16* __restrict__ Kt,  // [b,h,s,d]
                           const _Float16* __restrict__ Vt,  // [b,h,d,s]
                           _Float16* __restrict__ O) {       // [b,s,e]
  const int lane = threadIdx.x & 31;
  const int wv   = threadIdx.x >> 5;
  const int lan  = lane & 15;
  const int hf   = lane >> 4;
  const int id   = blockIdx.x * 4 + wv;        // 0 .. B*H*(S/16)-1
  const int qt   = id & 127;
  const int h    = (id >> 7) & 15;
  const int b    = id >> 11;
  const int q_base = qt * 16;

  const _Float16* qp = Q  + ((size_t)(b * HH + h) * SS) * DD;
  const _Float16* kp = Kt + ((size_t)(b * HH + h) * SS) * DD;
  const _Float16* vp = Vt + ((size_t)(b * HH + h) * DD) * SS;

  // B = Q^T: lane = query column, d interleaved across halves (8-blocks)
  v16h bq[2];
  #pragma unroll
  for (int c = 0; c < 2; c++) {
    const _Float16* p = qp + (size_t)(q_base + lan) * DD + c * 32 + hf * 8;
    bq[c] = cat8(*(const v8h*)p, *(const v8h*)(p + 16));
  }

  v8f zero = {0.f,0.f,0.f,0.f,0.f,0.f,0.f,0.f};
  v8f oacc[4] = {zero, zero, zero, zero};     // O^T d-tiles, col = query
  float mrun = -3.0e38f;
  float lrun = 0.f;
  const int qg = q_base + lan;                // my query (per lane, both halves)
  const int nj = q_base / 32 + 1;             // causal block count

  for (int j = 0; j < nj; j++) {
    const int key0 = j * 32;
    if (j + 1 < nj) {
      __builtin_prefetch(kp + (size_t)(key0 + 32) * DD, 0, 1);
      __builtin_prefetch(vp + (size_t)lane * SS + key0 + 32, 0, 1);
    }

    // A = K rows (16 keys x 64d per tile, two d-chunks of 32)
    v16h ak[2][2];
    #pragma unroll
    for (int kt = 0; kt < 2; kt++)
      #pragma unroll
      for (int c = 0; c < 2; c++) {
        const _Float16* p = kp + (size_t)(key0 + kt * 16 + lan) * DD + c * 32 + hf * 8;
        ak[kt][c] = cat8(*(const v8h*)p, *(const v8h*)(p + 16));
      }

    v8f s0 = zero, s1 = zero;                 // S^T tiles: rows=keys, cols=queries
    s0 = wmma_f16(ak[0][0], bq[0], s0);
    s0 = wmma_f16(ak[0][1], bq[1], s0);
    s1 = wmma_f16(ak[1][0], bq[0], s1);
    s1 = wmma_f16(ak[1][1], bq[1], s1);

    // scale 1/sqrt(64) + causal mask (key > query -> -1e10)
    const bool need_mask = (key0 + 31 > q_base);
    #pragma unroll
    for (int r = 0; r < 8; r++) {
      float x0 = s0[r] * 0.125f;
      float x1 = s1[r] * 0.125f;
      if (need_mask) {
        const int kg = key0 + r + 8 * hf;     // C: row = r + 8*half = key
        x0 = (kg      > qg) ? -1.0e10f : x0;
        x1 = (kg + 16 > qg) ? -1.0e10f : x1;
      }
      s0[r] = x0; s1[r] = x1;
    }

    // online softmax, per-query stats (query = lane column)
    float mb = -3.0e38f;
    #pragma unroll
    for (int r = 0; r < 8; r++) mb = fmaxf(mb, fmaxf(s0[r], s1[r]));
    mb = fmaxf(mb, __shfl_xor(mb, 16));       // combine key halves
    const float mn    = fmaxf(mrun, mb);
    const float alpha = __expf(mrun - mn);
    mrun = mn;

    float sum = 0.f;
    #pragma unroll
    for (int r = 0; r < 8; r++) {
      s0[r] = __expf(s0[r] - mn);
      s1[r] = __expf(s1[r] - mn);
      sum += s0[r] + s1[r];
    }
    sum += __shfl_xor(sum, 16);
    lrun = lrun * alpha + sum;

    // P^T as B operand: K(keys) split linearly across halves -> half swap
    v16h pb;
    #pragma unroll
    for (int r = 0; r < 8; r++) {
      const float t0 = __shfl_xor(s0[r], 16);
      const float t1 = __shfl_xor(s1[r], 16);
      pb[r]     = (_Float16)(hf ? t1 : s0[r]);  // K 0..7  | 16..23
      pb[r + 8] = (_Float16)(hf ? s1[r] : t0);  // K 8..15 | 24..31
    }

    // O^T += V^T @ P^T  (A = V^T rows, contiguous in key)
    #pragma unroll
    for (int t = 0; t < 4; t++) {
      const _Float16* p = vp + (size_t)(t * 16 + lan) * SS + key0 + hf * 16;
      v16h av = cat8(*(const v8h*)p, *(const v8h*)(p + 8));
      oacc[t] = oacc[t] * alpha;
      oacc[t] = wmma_f16(av, pb, oacc[t]);
    }
  }

  // normalize + store O[q, e] as f16 [b,s,e]; 8 contiguous d per tile/lane
  const float rl = 1.0f / lrun;
  #pragma unroll
  for (int t = 0; t < 4; t++) {
    v8h ov;
    #pragma unroll
    for (int r = 0; r < 8; r++) ov[r] = (_Float16)(oacc[t][r] * rl);
    _Float16* dst = O + ((size_t)b * SS + qg) * EE + h * DD + t * 16 + 8 * hf;
    *(v8h*)dst = ov;
  }
}

// ---------------------------------------------------------------------------
extern "C" void kernel_launch(void* const* d_in, const int* in_sizes, int n_in,
                              void* d_out, int out_size, void* d_ws, size_t ws_size,
                              hipStream_t stream) {
  (void)in_sizes; (void)n_in; (void)out_size; (void)ws_size;

  const float* xq = (const float*)d_in[0];
  const float* xk = (const float*)d_in[1];
  const float* xv = (const float*)d_in[2];
  // d_in[3] = mask: causal structure is applied analytically in flash_attn
  const float* Wq = (const float*)d_in[4];
  const float* bq = (const float*)d_in[5];
  const float* Wk = (const float*)d_in[6];
  const float* bk = (const float*)d_in[7];
  const float* Wv = (const float*)d_in[8];
  const float* bv = (const float*)d_in[9];
  const float* Wo = (const float*)d_in[10];
  const float* bo = (const float*)d_in[11];

  const size_t HSZ = (size_t)BB * HH * SS * DD;   // 4M f16 elements per tensor
  const size_t WSZ = (size_t)EE * EE;
  _Float16* q_ws = (_Float16*)d_ws;
  _Float16* k_ws = q_ws + HSZ;
  _Float16* v_ws = k_ws + HSZ;
  _Float16* o_ws = v_ws + HSZ;
  _Float16* wq16 = o_ws + HSZ;
  _Float16* wk16 = wq16 + WSZ;
  _Float16* wv16 = wk16 + WSZ;
  _Float16* wo16 = wv16 + WSZ;

  // one-shot weight conversion to f16 (amortized over M=4096)
  const int n4 = (int)(WSZ / 4);
  cvt_f32_f16<<<n4 / 256, 256, 0, stream>>>(Wq, wq16, n4);
  cvt_f32_f16<<<n4 / 256, 256, 0, stream>>>(Wk, wk16, n4);
  cvt_f32_f16<<<n4 / 256, 256, 0, stream>>>(Wv, wv16, n4);
  cvt_f32_f16<<<n4 / 256, 256, 0, stream>>>(Wo, wo16, n4);

  // projections: each wave owns a 16x64 tile; 4 waves/block
  dim3 gblk(EE / 64, MM / 64);
  gemm_qkvo<0, false><<<gblk, 128, 0, stream>>>(xq, wq16, bq, q_ws);
  gemm_qkvo<0, false><<<gblk, 128, 0, stream>>>(xk, wk16, bk, k_ws);
  gemm_qkvo<1, false><<<gblk, 128, 0, stream>>>(xv, wv16, bv, v_ws);

  // fused causal attention: B*H*(S/16) = 4096 waves, 4 per block
  flash_attn<<<(BB * HH * (SS / 16)) / 4, 128, 0, stream>>>(q_ws, k_ws, v_ws, o_ws);

  // output projection, fp32 result
  gemm_qkvo<2, true><<<gblk, 128, 0, stream>>>(o_ws, wo16, bo, d_out);
}
